// TransitionModel_31026843746378
// MI455X (gfx1250) — matrix operand order repaired and checked by reading, not compiled, not run on hardware
//
#include <hip/hip_runtime.h>
#include <hip/hip_bf16.h>

#define Bc  32
#define Tc  256
#define Dc  1024
#define Ac  64
#define Hc  512
#define NAc 20
#define G4D (4 * Dc)

typedef __bf16 bf16;
typedef __attribute__((ext_vector_type(16))) __bf16 v16bf;
typedef __attribute__((ext_vector_type(8)))  float  v8f;

__device__ __forceinline__ float sigmoidf_(float x) { return 1.0f / (1.0f + __expf(-x)); }

__device__ __forceinline__ v8f wmma_bf16(v16bf a, v16bf b, v8f c) {
  // D = A(16x32 bf16) x B(32x16 bf16) + C(16x16 f32)
  return __builtin_amdgcn_wmma_f32_16x16x32_bf16(false, a, false, b, (short)0, c, false, false);
}

// A fragment: 16(M) x 32(K) bf16, row-major source with leading dim ld.
// Lane L in [0,16): M=L, K = {8h..8h+7, 16+8h..16+8h+7} where h = lane>=16.
__device__ __forceinline__ v16bf load_fragA(const bf16* p, int ld) {
  const int lane = threadIdx.x & 31;
  const int r    = lane & 15;
  const int half = lane >> 4;
  const bf16* row = p + (size_t)r * ld + half * 8;
  union { uint4 u; bf16 h[8]; } lo, hi;
  lo.u = *(const uint4*)(row);
  hi.u = *(const uint4*)(row + 16);
  v16bf f;
#pragma unroll
  for (int i = 0; i < 8; ++i) { f[i] = lo.h[i]; f[i + 8] = hi.h[i]; }
  return f;
}

// B fragment: 32(K) x 16(N); stored N-major [N rows, K cols] row-major (i.e. W[n,k]).
// Lane holds N = lane&15, K = {8h..8h+7, 16+8h..16+8h+7} where h = lane>=16.
__device__ __forceinline__ v16bf load_fragB(const bf16* p, int ld) {
  const int lane = threadIdx.x & 31;
  const int r    = lane & 15;
  const int half = lane >> 4;
  const bf16* row = p + (size_t)r * ld + half * 8;
  union { uint4 u; bf16 h[8]; } lo, hi;
  lo.u = *(const uint4*)(row);
  hi.u = *(const uint4*)(row + 16);
  v16bf f;
#pragma unroll
  for (int i = 0; i < 8; ++i) { f[i] = lo.h[i]; f[i + 8] = hi.h[i]; }
  return f;
}

// ---------------------------------------------------------------- prep kernels

__global__ void cvt_kernel(bf16* dst, const float* src, size_t n) {
  size_t i = (size_t)blockIdx.x * blockDim.x + threadIdx.x;
  if (i < n) dst[i] = (bf16)src[i];
}

// red_W [4,D,2D] -> left half [4D,D], right half [4D,D], bf16
__global__ void red_split_kernel(bf16* dl, bf16* dr, const float* src) {
  size_t i = (size_t)blockIdx.x * blockDim.x + threadIdx.x;
  if (i < (size_t)G4D * Dc) {
    size_t row = i >> 10;            // 0..4095
    size_t k   = i & (Dc - 1);
    dl[i] = (bf16)src[row * (2 * Dc) + k];
    dr[i] = (bf16)src[row * (2 * Dc) + Dc + k];
  }
}

// emb_bf[(b*T+t)*A + a] = bf16(act_emb[actions[b*T+t]*A + a])
__global__ void gather_emb_kernel(bf16* dst, const int* actions, const float* emb) {
  int i = blockIdx.x * blockDim.x + threadIdx.x;
  if (i < Bc * Tc * Ac) {
    int row = i >> 6;
    int a   = i & (Ac - 1);
    dst[i] = (bf16)emb[actions[row] * Ac + a];
  }
}

// ---------------------------------------------------------------- batch GEMM
// C[M,N] (f32) = A[M,K](bf16) * Bm[N,K](bf16)^T + bias0[N] + bias1[N]
// block = 128 threads (4 waves); grid = (N/256, M/32).
// Each wave computes a 32x64 output tile: 2 A-frags x 4 B-frags -> 8 WMMA / K-chunk.
__global__ __launch_bounds__(128)
void gemm_bf16_kernel(const bf16* __restrict__ A, const bf16* __restrict__ Bm,
                      const float* __restrict__ bias0, const float* __restrict__ bias1,
                      float* __restrict__ C, int M, int N, int K) {
  const int wave = threadIdx.x >> 5;
  const int lane = threadIdx.x & 31;
  const int half = lane >> 4;
  const int nl   = lane & 15;
  const int n0 = blockIdx.x * 256 + wave * 64;
  const int m0 = blockIdx.y * 32;
  const int KC = K / 32;
  v8f acc[2][4];
#pragma unroll
  for (int mi = 0; mi < 2; ++mi)
#pragma unroll
    for (int j = 0; j < 4; ++j)
#pragma unroll
      for (int v = 0; v < 8; ++v) acc[mi][j][v] = 0.0f;

  for (int kc = 0; kc < KC; ++kc) {
    if (lane == 0 && kc + 4 < KC) {
      // stream-ahead hint for the weight matrix (global_prefetch_b8)
      __builtin_prefetch(Bm + (size_t)n0 * K + (kc + 4) * 32, 0, 1);
    }
    v16bf a0 = load_fragA(A + (size_t)m0 * K + kc * 32, K);
    v16bf a1 = load_fragA(A + (size_t)(m0 + 16) * K + kc * 32, K);
#pragma unroll
    for (int j = 0; j < 4; ++j) {
      v16bf b = load_fragB(Bm + (size_t)(n0 + j * 16) * K + kc * 32, K);
      acc[0][j] = wmma_bf16(a0, b, acc[0][j]);
      acc[1][j] = wmma_bf16(a1, b, acc[1][j]);
    }
  }
#pragma unroll
  for (int j = 0; j < 4; ++j) {
    const int n = n0 + j * 16 + nl;
    float bsum = (bias0 ? bias0[n] : 0.0f) + (bias1 ? bias1[n] : 0.0f);
#pragma unroll
    for (int mi = 0; mi < 2; ++mi)
#pragma unroll
      for (int v = 0; v < 8; ++v)
        C[(size_t)(m0 + mi * 16 + half * 8 + v) * N + n] = acc[mi][j][v] + bsum;
  }
}

// hid = sigmoid(opre) * tanh(cell); also bf16 copy
__global__ void leaf_combine_kernel(const float* __restrict__ cellp, const float* __restrict__ opre,
                                    float* __restrict__ hid, bf16* __restrict__ hid_bf, size_t n) {
  size_t i = (size_t)blockIdx.x * blockDim.x + threadIdx.x;
  if (i < n) {
    float h = sigmoidf_(opre[i]) * tanhf(cellp[i]);
    hid[i] = h;
    hid_bf[i] = (bf16)h;
  }
}

// ---------------------------------------------------------------- LSTM scan (persistent single WGP)
// gates[t] = h @ Whh^T + xg[b*T+t] (xg already holds x-proj + both biases)
__global__ __launch_bounds__(1024)
void lstm_scan_kernel(const float* __restrict__ xg, const bf16* __restrict__ Whh,
                      const float* __restrict__ h0, const float* __restrict__ c0,
                      float* s3_out, float* hf_out, bf16* hf_bf_out, float* cf_out) {
  __shared__ __align__(16) bf16 hbuf[Bc * Dc];  // 64 KB
  const int tid  = threadIdx.x;
  const int wave = tid >> 5;
  const int lane = tid & 31;
  const int half = lane >> 4;
  const int nl   = lane & 15;

  for (int i = tid; i < Bc * Dc; i += 1024) hbuf[i] = (bf16)h0[i & (Dc - 1)];
  float creg[2][2][8];
#pragma unroll
  for (int m = 0; m < 2; ++m)
#pragma unroll
    for (int eo = 0; eo < 2; ++eo) {
      int e = (wave * 2 + eo) * 16 + nl;
#pragma unroll
      for (int v = 0; v < 8; ++v) creg[m][eo][v] = c0[e];
    }
  __syncthreads();

  for (int t = 0; t < Tc; ++t) {
    v8f acc[2][2][4];
#pragma unroll
    for (int m = 0; m < 2; ++m)
#pragma unroll
      for (int eo = 0; eo < 2; ++eo)
#pragma unroll
        for (int g = 0; g < 4; ++g)
#pragma unroll
          for (int v = 0; v < 8; ++v) acc[m][eo][g][v] = 0.0f;

    for (int kc = 0; kc < Dc / 32; ++kc) {
      v16bf a0 = load_fragA(hbuf + kc * 32, Dc);
      v16bf a1 = load_fragA(hbuf + (size_t)16 * Dc + kc * 32, Dc);
#pragma unroll
      for (int eo = 0; eo < 2; ++eo) {
        int et = wave * 2 + eo;
#pragma unroll
        for (int g = 0; g < 4; ++g) {
          v16bf bfr = load_fragB(Whh + (size_t)(g * Dc + et * 16) * Dc + kc * 32, Dc);
          acc[0][eo][g] = wmma_bf16(a0, bfr, acc[0][eo][g]);
          acc[1][eo][g] = wmma_bf16(a1, bfr, acc[1][eo][g]);
        }
      }
    }
    __syncthreads();  // all reads of h done before overwrite

#pragma unroll
    for (int m = 0; m < 2; ++m) {
#pragma unroll
      for (int eo = 0; eo < 2; ++eo) {
        int e = (wave * 2 + eo) * 16 + nl;
#pragma unroll
        for (int v = 0; v < 8; ++v) {
          int bb = m * 16 + half * 8 + v;
          size_t row = ((size_t)bb * Tc + t) * G4D;
          float gi = acc[m][eo][0][v] + xg[row + 0 * Dc + e];
          float gf = acc[m][eo][1][v] + xg[row + 1 * Dc + e];
          float gg = acc[m][eo][2][v] + xg[row + 2 * Dc + e];
          float go = acc[m][eo][3][v] + xg[row + 3 * Dc + e];
          float c2 = sigmoidf_(gf) * creg[m][eo][v] + sigmoidf_(gi) * tanhf(gg);
          float h2 = sigmoidf_(go) * tanhf(c2);
          creg[m][eo][v] = c2;
          hbuf[bb * Dc + e] = (bf16)h2;
          if (s3_out && t == Tc - 2) s3_out[(size_t)bb * Dc + e] = h2;
          if (t == Tc - 1) {
            hf_out[(size_t)bb * Dc + e] = h2;
            if (hf_bf_out) hf_bf_out[(size_t)bb * Dc + e] = (bf16)h2;
            if (cf_out)    cf_out[(size_t)bb * Dc + e] = c2;
          }
        }
      }
    }
    __syncthreads();  // h fully written before next step reads
  }
}

// ---------------------------------------------------------------- reduce tree-LSTM scan
// gates[t] = lh @ Wl^T + rg[b*T+t]   (rg holds rh-proj + red_b)
__global__ __launch_bounds__(1024)
void reduce_scan_kernel(const float* __restrict__ rg, const bf16* __restrict__ Wl,
                        const float* __restrict__ hid, const float* __restrict__ cellp,
                        bf16* __restrict__ redh_bf) {
  __shared__ __align__(16) bf16 hbuf[Bc * Dc];
  const int tid  = threadIdx.x;
  const int wave = tid >> 5;
  const int lane = tid & 31;
  const int half = lane >> 4;
  const int nl   = lane & 15;

  for (int i = tid; i < Bc * Dc; i += 1024)
    hbuf[i] = (bf16)hid[((size_t)(i >> 10) * Tc) * Dc + (i & (Dc - 1))];  // hid_t[0]
  float creg[2][2][8];
#pragma unroll
  for (int m = 0; m < 2; ++m)
#pragma unroll
    for (int eo = 0; eo < 2; ++eo) {
      int e = (wave * 2 + eo) * 16 + nl;
#pragma unroll
      for (int v = 0; v < 8; ++v) {
        int bb = m * 16 + half * 8 + v;
        creg[m][eo][v] = cellp[((size_t)bb * Tc) * Dc + e];  // cell_t[0]
      }
    }
  __syncthreads();

  for (int t = 1; t < Tc; ++t) {
    v8f acc[2][2][4];
#pragma unroll
    for (int m = 0; m < 2; ++m)
#pragma unroll
      for (int eo = 0; eo < 2; ++eo)
#pragma unroll
        for (int g = 0; g < 4; ++g)
#pragma unroll
          for (int v = 0; v < 8; ++v) acc[m][eo][g][v] = 0.0f;

    for (int kc = 0; kc < Dc / 32; ++kc) {
      v16bf a0 = load_fragA(hbuf + kc * 32, Dc);
      v16bf a1 = load_fragA(hbuf + (size_t)16 * Dc + kc * 32, Dc);
#pragma unroll
      for (int eo = 0; eo < 2; ++eo) {
        int et = wave * 2 + eo;
#pragma unroll
        for (int g = 0; g < 4; ++g) {
          v16bf bfr = load_fragB(Wl + (size_t)(g * Dc + et * 16) * Dc + kc * 32, Dc);
          acc[0][eo][g] = wmma_bf16(a0, bfr, acc[0][eo][g]);
          acc[1][eo][g] = wmma_bf16(a1, bfr, acc[1][eo][g]);
        }
      }
    }
    __syncthreads();

#pragma unroll
    for (int m = 0; m < 2; ++m) {
#pragma unroll
      for (int eo = 0; eo < 2; ++eo) {
        int e = (wave * 2 + eo) * 16 + nl;
#pragma unroll
        for (int v = 0; v < 8; ++v) {
          int bb = m * 16 + half * 8 + v;
          size_t row = ((size_t)bb * Tc + t) * G4D;
          float gi = acc[m][eo][0][v] + rg[row + 0 * Dc + e];
          float gl = acc[m][eo][1][v] + rg[row + 1 * Dc + e];
          float gr = acc[m][eo][2][v] + rg[row + 2 * Dc + e];
          float gc = acc[m][eo][3][v] + rg[row + 3 * Dc + e];
          float rc = cellp[((size_t)bb * Tc + t) * Dc + e];
          float c2 = sigmoidf_(gi) * tanhf(gc) + sigmoidf_(gl) * creg[m][eo][v] + sigmoidf_(gr) * rc;
          float h2 = tanhf(c2);
          creg[m][eo][v] = c2;
          hbuf[bb * Dc + e] = (bf16)h2;
          if (t == Tc - 1) redh_bf[(size_t)bb * Dc + e] = (bf16)h2;
        }
      }
    }
    __syncthreads();
  }
}

// ---------------------------------------------------------------- one extra LSTM cell (s1)
// gates = xh@Wih^T + hh@Whh^T + bih + bhh; combine with cprev
__global__ __launch_bounds__(1024)
void final_cell_kernel(const bf16* __restrict__ xh, const bf16* __restrict__ Wih,
                       const bf16* __restrict__ hh, const bf16* __restrict__ Whh,
                       const float* __restrict__ bih, const float* __restrict__ bhh,
                       const float* __restrict__ cprev, float* __restrict__ h_out) {
  const int wave = threadIdx.x >> 5;
  const int lane = threadIdx.x & 31;
  const int half = lane >> 4;
  const int nl   = lane & 15;
  v8f acc[2][2][4];
#pragma unroll
  for (int m = 0; m < 2; ++m)
#pragma unroll
    for (int eo = 0; eo < 2; ++eo)
#pragma unroll
      for (int g = 0; g < 4; ++g)
#pragma unroll
        for (int v = 0; v < 8; ++v) acc[m][eo][g][v] = 0.0f;

  for (int kc = 0; kc < Dc / 32; ++kc) {
    v16bf a0 = load_fragA(xh + kc * 32, Dc);
    v16bf a1 = load_fragA(xh + (size_t)16 * Dc + kc * 32, Dc);
#pragma unroll
    for (int eo = 0; eo < 2; ++eo) {
      int et = wave * 2 + eo;
#pragma unroll
      for (int g = 0; g < 4; ++g) {
        v16bf bfr = load_fragB(Wih + (size_t)(g * Dc + et * 16) * Dc + kc * 32, Dc);
        acc[0][eo][g] = wmma_bf16(a0, bfr, acc[0][eo][g]);
        acc[1][eo][g] = wmma_bf16(a1, bfr, acc[1][eo][g]);
      }
    }
  }
  for (int kc = 0; kc < Dc / 32; ++kc) {
    v16bf a0 = load_fragA(hh + kc * 32, Dc);
    v16bf a1 = load_fragA(hh + (size_t)16 * Dc + kc * 32, Dc);
#pragma unroll
    for (int eo = 0; eo < 2; ++eo) {
      int et = wave * 2 + eo;
#pragma unroll
      for (int g = 0; g < 4; ++g) {
        v16bf bfr = load_fragB(Whh + (size_t)(g * Dc + et * 16) * Dc + kc * 32, Dc);
        acc[0][eo][g] = wmma_bf16(a0, bfr, acc[0][eo][g]);
        acc[1][eo][g] = wmma_bf16(a1, bfr, acc[1][eo][g]);
      }
    }
  }
#pragma unroll
  for (int m = 0; m < 2; ++m) {
#pragma unroll
    for (int eo = 0; eo < 2; ++eo) {
      int e = (wave * 2 + eo) * 16 + nl;
#pragma unroll
      for (int v = 0; v < 8; ++v) {
        int bb = m * 16 + half * 8 + v;
        float gi = acc[m][eo][0][v] + bih[0 * Dc + e] + bhh[0 * Dc + e];
        float gf = acc[m][eo][1][v] + bih[1 * Dc + e] + bhh[1 * Dc + e];
        float gg = acc[m][eo][2][v] + bih[2 * Dc + e] + bhh[2 * Dc + e];
        float go = acc[m][eo][3][v] + bih[3 * Dc + e] + bhh[3 * Dc + e];
        float c2 = sigmoidf_(gf) * cprev[(size_t)bb * Dc + e] + sigmoidf_(gi) * tanhf(gg);
        h_out[(size_t)bb * Dc + e] = sigmoidf_(go) * tanhf(c2);
      }
    }
  }
}

// ---------------------------------------------------------------- attention (f32, small)

// u[j,b,e] = sum_d s_j[b,d] * attW[j][d,e]
__global__ void att_u_kernel(const float* __restrict__ s1, const float* __restrict__ s2,
                             const float* __restrict__ s3, const float* __restrict__ attW,
                             float* __restrict__ u) {
  int e = blockIdx.x * blockDim.x + threadIdx.x;
  int b = blockIdx.y, j = blockIdx.z;
  const float* s = (j == 0) ? s1 : ((j == 1) ? s2 : s3);
  const float* W = attW + (size_t)j * Dc * Dc;
  float acc = 0.0f;
  for (int d = 0; d < Dc; ++d) acc += s[(size_t)b * Dc + d] * W[(size_t)d * Dc + e];
  u[((size_t)j * Bc + b) * Dc + e] = acc;
}

// scores -> softmax over T -> weighted sum of sentence
__global__ __launch_bounds__(256)
void att_apply_kernel(const float* __restrict__ u, const float* __restrict__ sent,
                      float* __restrict__ att) {
  __shared__ float wsh[Tc];
  __shared__ float redb[256];
  int t = threadIdx.x, b = blockIdx.x, j = blockIdx.y;
  const float* ub = u + ((size_t)j * Bc + b) * Dc;
  const float* sb = sent + (size_t)b * Tc * Dc;
  float sc = 0.0f;
  for (int d = 0; d < Dc; ++d) sc += ub[d] * sb[(size_t)t * Dc + d];
  redb[t] = sc;
  __syncthreads();
  for (int s = 128; s > 0; s >>= 1) { if (t < s) redb[t] = fmaxf(redb[t], redb[t + s]); __syncthreads(); }
  float mx = redb[0];
  __syncthreads();
  float ex = __expf(sc - mx);
  redb[t] = ex;
  __syncthreads();
  for (int s = 128; s > 0; s >>= 1) { if (t < s) redb[t] += redb[t + s]; __syncthreads(); }
  wsh[t] = ex / redb[0];
  __syncthreads();
  for (int d = t; d < Dc; d += 256) {
    float a = 0.0f;
    for (int tt = 0; tt < Tc; ++tt) a += wsh[tt] * sb[(size_t)tt * Dc + d];
    att[((size_t)j * Bc + b) * Dc + d] = a;
  }
}

// ---------------------------------------------------------------- head (f32, small)

__global__ void hidden_kernel(const float* __restrict__ s1, const float* __restrict__ s2,
                              const float* __restrict__ s3, const float* __restrict__ tok,
                              const float* __restrict__ atop, const float* __restrict__ att,
                              const float* __restrict__ W, const float* __restrict__ bias,
                              float* __restrict__ hidden) {
  int idx = blockIdx.x * blockDim.x + threadIdx.x;
  if (idx >= Bc * Hc) return;
  int b = idx / Hc, h = idx % Hc;
  const float* srcs[8] = { s1 + (size_t)b * Dc, s2 + (size_t)b * Dc, s3 + (size_t)b * Dc,
                           tok, atop + (size_t)b * Dc,
                           att + ((size_t)0 * Bc + b) * Dc, att + ((size_t)1 * Bc + b) * Dc,
                           att + ((size_t)2 * Bc + b) * Dc };
  float acc = bias[h];
  const float* wr = W + (size_t)h * 8 * Dc;
  for (int seg = 0; seg < 8; ++seg) {
    const float* s = srcs[seg];
    const float* wseg = wr + (size_t)seg * Dc;
    for (int e = 0; e < Dc; ++e) acc += s[e] * wseg[e];
  }
  hidden[idx] = tanhf(acc);
}

__global__ void logits_kernel(const float* __restrict__ hidden, const float* __restrict__ W,
                              const float* __restrict__ bias, float* __restrict__ out) {
  int b = threadIdx.x;
  if (b >= Bc) return;
  float l[NAc];
  float mx = -1e30f;
  for (int n = 0; n < NAc; ++n) {
    float a = bias[n];
    for (int h = 0; h < Hc; ++h) a += hidden[(size_t)b * Hc + h] * W[(size_t)n * Hc + h];
    l[n] = a;
    mx = fmaxf(mx, a);
  }
  float s = 0.0f;
  for (int n = 0; n < NAc; ++n) s += __expf(l[n] - mx);
  float lz = mx + __logf(s);
  for (int n = 0; n < NAc; ++n) out[(size_t)b * NAc + n] = l[n] - lz;
}

// ---------------------------------------------------------------- launch

extern "C" void kernel_launch(void* const* d_in, const int* in_sizes, int n_in,
                              void* d_out, int out_size, void* d_ws, size_t ws_size,
                              hipStream_t stream) {
  (void)in_sizes; (void)n_in; (void)out_size; (void)ws_size;
  const float* sentence    = (const float*)d_in[0];
  const int*   actions     = (const int*)  d_in[1];
  const float* token_empty = (const float*)d_in[2];
  const float* leaf_Wi     = (const float*)d_in[3];
  const float* leaf_bi     = (const float*)d_in[4];
  const float* leaf_Wo     = (const float*)d_in[5];
  const float* leaf_bo     = (const float*)d_in[6];
  const float* lstm_Wih    = (const float*)d_in[7];
  const float* lstm_Whh    = (const float*)d_in[8];
  const float* lstm_bih    = (const float*)d_in[9];
  const float* lstm_bhh    = (const float*)d_in[10];
  const float* stack_h0    = (const float*)d_in[11];
  const float* stack_c0    = (const float*)d_in[12];
  const float* act_emb     = (const float*)d_in[13];
  const float* act_Wih     = (const float*)d_in[14];
  const float* act_Whh     = (const float*)d_in[15];
  const float* act_bih     = (const float*)d_in[16];
  const float* act_bhh     = (const float*)d_in[17];
  const float* act_h0      = (const float*)d_in[18];
  const float* act_c0      = (const float*)d_in[19];
  const float* red_W       = (const float*)d_in[20];
  const float* red_b       = (const float*)d_in[21];
  const float* h2f_W       = (const float*)d_in[22];
  const float* h2f_b       = (const float*)d_in[23];
  const float* f2a_W       = (const float*)d_in[24];
  const float* f2a_b       = (const float*)d_in[25];
  const float* att_W       = (const float*)d_in[26];

  char* base = (char*)d_ws;
  size_t off = 0;
  auto alloc = [&](size_t bytes) -> void* {
    void* p = base + off;
    off += (bytes + 255) & ~(size_t)255;
    return p;
  };
  bf16*  sent_bf    = (bf16*) alloc((size_t)Bc * Tc * Dc * 2);
  bf16*  leafWi_bf  = (bf16*) alloc((size_t)Dc * Dc * 2);
  bf16*  leafWo_bf  = (bf16*) alloc((size_t)Dc * Dc * 2);
  bf16*  lstmWih_bf = (bf16*) alloc((size_t)G4D * Dc * 2);
  bf16*  lstmWhh_bf = (bf16*) alloc((size_t)G4D * Dc * 2);
  bf16*  actWih_bf  = (bf16*) alloc((size_t)G4D * Ac * 2);
  bf16*  actWhh_bf  = (bf16*) alloc((size_t)G4D * Dc * 2);
  bf16*  redWl_bf   = (bf16*) alloc((size_t)G4D * Dc * 2);
  bf16*  redWr_bf   = (bf16*) alloc((size_t)G4D * Dc * 2);
  bf16*  emb_bf     = (bf16*) alloc((size_t)Bc * Tc * Ac * 2);
  float* cellbuf    = (float*)alloc((size_t)Bc * Tc * Dc * 4);
  float* opre       = (float*)alloc((size_t)Bc * Tc * Dc * 4);
  float* hid        = (float*)alloc((size_t)Bc * Tc * Dc * 4);
  bf16*  hid_bf     = (bf16*) alloc((size_t)Bc * Tc * Dc * 2);
  float* xg         = (float*)alloc((size_t)Bc * Tc * G4D * 4);
  float* ax         = (float*)alloc((size_t)Bc * Tc * G4D * 4);
  float* rg         = (float*)alloc((size_t)Bc * Tc * G4D * 4);
  float* s3buf      = (float*)alloc((size_t)Bc * Dc * 4);
  float* hfbuf      = (float*)alloc((size_t)Bc * Dc * 4);
  bf16*  hf_bf      = (bf16*) alloc((size_t)Bc * Dc * 2);
  float* cfbuf      = (float*)alloc((size_t)Bc * Dc * 4);
  float* atop       = (float*)alloc((size_t)Bc * Dc * 4);
  bf16*  redh_bf    = (bf16*) alloc((size_t)Bc * Dc * 2);
  float* s1buf      = (float*)alloc((size_t)Bc * Dc * 4);
  float* ubuf       = (float*)alloc((size_t)3 * Bc * Dc * 4);
  float* attbuf     = (float*)alloc((size_t)3 * Bc * Dc * 4);
  float* hiddenb    = (float*)alloc((size_t)Bc * Hc * 4);

  auto cvt = [&](bf16* dst, const float* src, size_t n) {
    cvt_kernel<<<dim3((unsigned)((n + 255) / 256)), dim3(256), 0, stream>>>(dst, src, n);
  };
  cvt(sent_bf, sentence, (size_t)Bc * Tc * Dc);
  cvt(leafWi_bf, leaf_Wi, (size_t)Dc * Dc);
  cvt(leafWo_bf, leaf_Wo, (size_t)Dc * Dc);
  cvt(lstmWih_bf, lstm_Wih, (size_t)G4D * Dc);
  cvt(lstmWhh_bf, lstm_Whh, (size_t)G4D * Dc);
  cvt(actWih_bf, act_Wih, (size_t)G4D * Ac);
  cvt(actWhh_bf, act_Whh, (size_t)G4D * Dc);
  red_split_kernel<<<dim3((G4D * Dc + 255) / 256), dim3(256), 0, stream>>>(redWl_bf, redWr_bf, red_W);
  gather_emb_kernel<<<dim3((Bc * Tc * Ac + 255) / 256), dim3(256), 0, stream>>>(emb_bf, actions, act_emb);

  // leaf module (batch GEMMs): each block 32 rows x 256 cols
  gemm_bf16_kernel<<<dim3(Dc / 256, Bc * Tc / 32), dim3(128), 0, stream>>>(
      sent_bf, leafWi_bf, leaf_bi, nullptr, cellbuf, Bc * Tc, Dc, Dc);
  gemm_bf16_kernel<<<dim3(Dc / 256, Bc * Tc / 32), dim3(128), 0, stream>>>(
      sent_bf, leafWo_bf, leaf_bo, nullptr, opre, Bc * Tc, Dc, Dc);
  leaf_combine_kernel<<<dim3((Bc * Tc * Dc) / 256), dim3(256), 0, stream>>>(
      cellbuf, opre, hid, hid_bf, (size_t)Bc * Tc * Dc);

  // hoisted x-projections (batch GEMMs, biases folded)
  gemm_bf16_kernel<<<dim3(G4D / 256, Bc * Tc / 32), dim3(128), 0, stream>>>(
      hid_bf, lstmWih_bf, lstm_bih, lstm_bhh, xg, Bc * Tc, G4D, Dc);
  gemm_bf16_kernel<<<dim3(G4D / 256, Bc * Tc / 32), dim3(128), 0, stream>>>(
      emb_bf, actWih_bf, act_bih, act_bhh, ax, Bc * Tc, G4D, Ac);
  gemm_bf16_kernel<<<dim3(G4D / 256, Bc * Tc / 32), dim3(128), 0, stream>>>(
      hid_bf, redWr_bf, red_b, nullptr, rg, Bc * Tc, G4D, Dc);

  // sequential recurrences (persistent single-WGP WMMA scans)
  lstm_scan_kernel<<<dim3(1), dim3(1024), 0, stream>>>(
      xg, lstmWhh_bf, stack_h0, stack_c0, s3buf, hfbuf, hf_bf, cfbuf);
  lstm_scan_kernel<<<dim3(1), dim3(1024), 0, stream>>>(
      ax, actWhh_bf, act_h0, act_c0, nullptr, atop, nullptr, nullptr);
  reduce_scan_kernel<<<dim3(1), dim3(1024), 0, stream>>>(
      rg, redWl_bf, hid, cellbuf, redh_bf);
  final_cell_kernel<<<dim3(1), dim3(1024), 0, stream>>>(
      redh_bf, lstmWih_bf, hf_bf, lstmWhh_bf, lstm_bih, lstm_bhh, cfbuf, s1buf);

  // attention over sentence for s1, s2(=hf), s3
  att_u_kernel<<<dim3(Dc / 256, Bc, 3), dim3(256), 0, stream>>>(s1buf, hfbuf, s3buf, att_W, ubuf);
  att_apply_kernel<<<dim3(Bc, 3), dim3(256), 0, stream>>>(ubuf, sentence, attbuf);

  // head
  hidden_kernel<<<dim3((Bc * Hc) / 256), dim3(256), 0, stream>>>(
      s1buf, hfbuf, s3buf, token_empty, atop, attbuf, h2f_W, h2f_b, hiddenb);
  logits_kernel<<<dim3(1), dim3(32), 0, stream>>>(hiddenb, f2a_W, f2a_b, (float*)d_out);
}
